// FeatureLine_309237645366
// MI455X (gfx1250) — matrix operand order repaired and checked by reading, not compiled, not run on hardware
//
#include <hip/hip_runtime.h>
#include <hip/hip_bf16.h>

typedef __attribute__((ext_vector_type(16))) _Float16 v16h;
typedef __attribute__((ext_vector_type(8)))  _Float16 v8h;
typedef __attribute__((ext_vector_type(8)))  float    v8f;

#define EXPRN 80
#define LLEN  64
#define CCH   32

// ---------------- Kernel 1: contract 96 feature lines into six [64][32] f16 tables
__global__ void tab_kernel(const float* __restrict__ expr, const float* __restrict__ jw,
                           const float* __restrict__ flx, const float* __restrict__ fly,
                           const float* __restrict__ flz, _Float16* __restrict__ tab) {
    int gid = blockIdx.x * blockDim.x + threadIdx.x;   // 0..12287
    int t  = gid >> 11;          // table index 0..5
    int lc = gid & 2047;         // l*32 + c
    int ax = t % 3;
    const float* src = (ax == 0) ? flx : (ax == 1) ? fly : flz;
    float s = 0.f;
    if (t < 3) {
        for (int i = 0; i < EXPRN; i++) s += src[i * 2048 + lc] * expr[i];
    } else {
        for (int i = 0; i < 16; i++)    s += src[(EXPRN + i) * 2048 + lc] * jw[i];
    }
    tab[gid] = (_Float16)s;
}

// ---------------- Kernel 2: weight-norm fold + pack to CDNA5 B-fragment layout
// B fragment for (nt, kc): lane%16 = N col nt*16+lane%16; element e = K row kc*32+16*(e/8)+8*(lane/16)+e%8.
__device__ void pack_wB(const float* __restrict__ v, const float* __restrict__ g,
                        _Float16* __restrict__ out, int K, int KC, float* s_scale) {
    int t = threadIdx.x;
    if (t < 128) {
        float s = 0.f;
        for (int k = 0; k < K; k++) { float x = v[t * K + k]; s += x * x; }
        s_scale[t] = g[t] * rsqrtf(s);
    }
    __syncthreads();
    int total = 8 * KC * 32 * 16;
    for (int f = t; f < total; f += blockDim.x) {
        int e    = f & 15;
        int lane = (f >> 4) & 31;
        int blk  = f >> 9;            // nt*KC + kc
        int kc   = blk % KC;
        int nt   = blk / KC;
        int n = nt * 16 + (lane & 15);
        int k = kc * 32 + (e >> 3) * 16 + (lane >> 4) * 8 + (e & 7);
        out[f] = (_Float16)(v[n * K + k] * s_scale[n]);
    }
}

__global__ void prep_kernel(const float* __restrict__ w0v, const float* __restrict__ w0g,
                            const float* __restrict__ w1v, const float* __restrict__ w1g,
                            const float* __restrict__ w2v, const float* __restrict__ w2g,
                            _Float16* __restrict__ W0B, _Float16* __restrict__ W1B,
                            float* __restrict__ w2f) {
    __shared__ float s_scale[128];
    if (blockIdx.x == 0) {
        pack_wB(w0v, w0g, W0B, 192, 6, s_scale);
    } else if (blockIdx.x == 1) {
        pack_wB(w1v, w1g, W1B, 128, 4, s_scale);
    } else {
        int t = threadIdx.x;
        if (t == 0) {
            float s = 0.f;
            for (int k = 0; k < 128; k++) s += w2v[k] * w2v[k];
            s_scale[0] = w2g[0] * rsqrtf(s);
        }
        __syncthreads();
        if (t < 128) w2f[t] = w2v[t] * s_scale[0];
    }
}

// ---------------- Kernel 3: main — features + 2 WMMA layers + final dot
__global__ void __launch_bounds__(128)
mlp_kernel(const float* __restrict__ xyz, const _Float16* __restrict__ g_tab,
           const _Float16* __restrict__ W0B, const _Float16* __restrict__ W1B,
           const float* __restrict__ w2f, const float* __restrict__ b0,
           const float* __restrict__ b1, const float* __restrict__ b2,
           float* __restrict__ out, int n) {
    __shared__ __align__(32) _Float16 s_tab[6 * LLEN * CCH];     // 24 KB
    __shared__ __align__(32) _Float16 s_scr[4][16 * 192];        // 24 KB, per-wave scratch

    int tid  = threadIdx.x;
    int wave = tid >> 5;
    int lane = tid & 31;
    int half = lane >> 4;        // 0: K-lo half rows, 1: K-hi half rows
    int lm   = lane & 15;

    // cooperative load of the 6 tables into LDS (v8h = b128 copies)
    for (int i = tid; i < (6 * LLEN * CCH) / 8; i += blockDim.x)
        ((v8h*)s_tab)[i] = ((const v8h*)g_tab)[i];
    __syncthreads();

    // per-lane biases in C/D layout: lane holds column nt*16+lm
    float bias0[8], bias1[8];
#pragma unroll
    for (int nt = 0; nt < 8; nt++) {
        bias0[nt] = b0[nt * 16 + lm];
        bias1[nt] = b1[nt * 16 + lm];
    }
    float b2s = b2[0];

    _Float16* scr = s_scr[wave];
    int koff = half * 8;
    int tiles = (n + 15) >> 4;
    int stride4 = gridDim.x * 4;
    int tile = blockIdx.x * 4 + wave;

    // software pipeline: preload this tile's coordinates
    float cx, cy, cz;
    {
        int pidx = tile * 16 + lm;
        if (pidx >= n) pidx = n - 1;
        cx = xyz[pidx * 3 + 0];
        cy = xyz[pidx * 3 + 1];
        cz = xyz[pidx * 3 + 2];
    }

    for (; tile < tiles; tile += stride4) {
        int base = tile * 16;

        // issue next tile's coordinate loads NOW; consumed at end of iteration,
        // so the global-load latency overlaps the whole WMMA section (1 wave/SIMD).
        int npidx = (tile + stride4) * 16 + lm;
        if (npidx >= n) npidx = n - 1;
        float nx = xyz[npidx * 3 + 0];
        float ny = xyz[npidx * 3 + 1];
        float nz = xyz[npidx * 3 + 2];

        float coords[3] = {cx, cy, cz};

        // -------- features: lane handles point lm; half selects segments 0-2 / 3-5
#pragma unroll
        for (int ss = 0; ss < 3; ss++) {
            int s = half * 3 + ss;                 // segment/table index; coord = s%3 = ss
            float c = coords[ss];
            float p = fminf(fmaxf(c, 0.f), 1.f) * (float)(LLEN - 1);
            int li = (int)p; if (li > LLEN - 1) li = LLEN - 1;
            float w = p - (float)li;
            int ri = li + 1; if (ri > LLEN - 1) ri = LLEN - 1;
            const v8h* lo = (const v8h*)&s_tab[(s * LLEN + li) * CCH];
            const v8h* hi = (const v8h*)&s_tab[(s * LLEN + ri) * CCH];
            v8h* dst = (v8h*)&scr[lm * 192 + s * 32];
#pragma unroll
            for (int g = 0; g < 4; g++) {
                v8h L = lo[g], H = hi[g], R;
#pragma unroll
                for (int e = 0; e < 8; e++) {
                    float lv = (float)L[e];
                    R[e] = (_Float16)(lv + w * ((float)H[e] - lv));
                }
                dst[g] = R;
            }
        }
        // per-wave scratch; same-wave DS ops are in-order, no barrier needed

        // -------- layer 0: [16x192] @ [192x128], 6 K-chunks x 8 N-tiles
        v8f acc0[8];
#pragma unroll
        for (int i = 0; i < 8; i++)
#pragma unroll
            for (int j = 0; j < 8; j++) acc0[i][j] = 0.f;

#pragma unroll
        for (int kc = 0; kc < 6; kc++) {
            union { v16h v; v8h h[2]; } A;
            const _Float16* ap = &scr[lm * 192 + kc * 32 + koff];
            A.h[0] = *(const v8h*)ap;
            A.h[1] = *(const v8h*)(ap + 16);
#pragma unroll
            for (int nt = 0; nt < 8; nt++) {
                v16h B = *(const v16h*)&W0B[((nt * 6 + kc) * 32 + lane) * 16];
                acc0[nt] = __builtin_amdgcn_wmma_f32_16x16x32_f16(
                    false, A.v, false, B, (short)0, acc0[nt], false, false);
            }
        }

        // bias + relu, write h0 as f16 tile [16][128] into scratch (A-gather layout)
        int mrow = half * 8;     // C/D layout: lane half selects M rows j / j+8
#pragma unroll
        for (int nt = 0; nt < 8; nt++)
#pragma unroll
            for (int j = 0; j < 8; j++) {
                float v = acc0[nt][j] + bias0[nt];
                v = v > 0.f ? v : 0.f;
                scr[(mrow + j) * 128 + nt * 16 + lm] = (_Float16)v;
            }

        // -------- layer 1: [16x128] @ [128x128], 4 K-chunks x 8 N-tiles
        v8f acc1[8];
#pragma unroll
        for (int i = 0; i < 8; i++)
#pragma unroll
            for (int j = 0; j < 8; j++) acc1[i][j] = 0.f;

#pragma unroll
        for (int kc = 0; kc < 4; kc++) {
            union { v16h v; v8h h[2]; } A;
            const _Float16* ap = &scr[lm * 128 + kc * 32 + koff];
            A.h[0] = *(const v8h*)ap;
            A.h[1] = *(const v8h*)(ap + 16);
#pragma unroll
            for (int nt = 0; nt < 8; nt++) {
                v16h B = *(const v16h*)&W1B[((nt * 4 + kc) * 32 + lane) * 16];
                acc1[nt] = __builtin_amdgcn_wmma_f32_16x16x32_f16(
                    false, A.v, false, B, (short)0, acc1[nt], false, false);
            }
        }

        // bias + relu, write h1 f16 tile back to scratch
#pragma unroll
        for (int nt = 0; nt < 8; nt++)
#pragma unroll
            for (int j = 0; j < 8; j++) {
                float v = acc1[nt][j] + bias1[nt];
                v = v > 0.f ? v : 0.f;
                scr[(mrow + j) * 128 + nt * 16 + lm] = (_Float16)v;
            }

        // -------- layer 2: 128 -> 1 dot; 2 lanes per point, pair-shuffle reduce
        int p2 = lane >> 1;
        int c0 = (lane & 1) * 64;
        float s2 = 0.f;
        const v8h* hrow = (const v8h*)&scr[p2 * 128 + c0];
#pragma unroll
        for (int g = 0; g < 8; g++) {
            v8h h = hrow[g];
#pragma unroll
            for (int e = 0; e < 8; e++)
                s2 += (float)h[e] * w2f[c0 + g * 8 + e];
        }
        s2 += __shfl_xor(s2, 1);
        if ((lane & 1) == 0 && base + p2 < n)
            out[base + p2] = s2 + b2s;

        // rotate pipeline registers
        cx = nx; cy = ny; cz = nz;
    }
}

extern "C" void kernel_launch(void* const* d_in, const int* in_sizes, int n_in,
                              void* d_out, int out_size, void* d_ws, size_t ws_size,
                              hipStream_t stream) {
    const float* expr = (const float*)d_in[0];
    const float* jw   = (const float*)d_in[1];
    const float* xyz  = (const float*)d_in[2];
    const float* flx  = (const float*)d_in[3];
    const float* fly  = (const float*)d_in[4];
    const float* flz  = (const float*)d_in[5];
    const float* w0v  = (const float*)d_in[6];
    const float* w0g  = (const float*)d_in[7];
    const float* b0   = (const float*)d_in[8];
    const float* w1v  = (const float*)d_in[9];
    const float* w1g  = (const float*)d_in[10];
    const float* b1   = (const float*)d_in[11];
    const float* w2v  = (const float*)d_in[12];
    const float* w2g  = (const float*)d_in[13];
    const float* b2   = (const float*)d_in[14];
    int n = in_sizes[2] / 3;

    char* ws = (char*)d_ws;
    _Float16* g_tab = (_Float16*)(ws);                            // 24576 B
    _Float16* W0B   = (_Float16*)(ws + 24576);                    // 49152 B
    _Float16* W1B   = (_Float16*)(ws + 24576 + 49152);            // 32768 B
    float*    w2f   = (float*)   (ws + 24576 + 49152 + 32768);    //   512 B

    tab_kernel<<<48, 256, 0, stream>>>(expr, jw, flx, fly, flz, g_tab);
    prep_kernel<<<3, 256, 0, stream>>>(w0v, w0g, w1v, w1g, w2v, w2g, W0B, W1B, w2f);
    mlp_kernel<<<1024, 128, 0, stream>>>(xyz, g_tab, W0B, W1B, w2f, b0, b1, b2,
                                         (float*)d_out, n);
}
